// Transformer_sstrack_29386166239727
// MI455X (gfx1250) — matrix operand order repaired and checked by reading, not compiled, not run on hardware
//
#include <hip/hip_runtime.h>
#include <cstdint>
#include <cstddef>

// ============================================================================
// CDNA5 / gfx1250 implementation.
// All GEMM-shaped work lowers to v_wmma_f32_16x16x32_bf16 (wave32 WMMA).
// f32 master copies live in d_ws; bf16 conversion happens at LDS staging
// (packed pairs -> v_cvt_pk_bf16_f32 + ds_store_b32).
// Dead subgraph icmp1 (the `_unused = icmp(...)` call) is eliminated.
// Transpose tiles use gfx1250 async global->LDS loads when available.
// ============================================================================

typedef __attribute__((ext_vector_type(16))) __bf16 v16bf;
typedef __attribute__((ext_vector_type(8)))  __bf16 v8bf;
typedef __attribute__((ext_vector_type(8)))  float  v8f;

#define BM 128
#define BN 64
#define BK 64   // two WMMA K-steps per LDS stage

// ---- gfx1250 async global->LDS support (guarded: absence cannot break build)
#if defined(__gfx1250__) && __has_builtin(__builtin_amdgcn_global_load_async_to_lds_b32)
#define HAVE_ASYNC_LDS 1
#else
#define HAVE_ASYNC_LDS 0
#endif

#if HAVE_ASYNC_LDS
__device__ __forceinline__ void async_load_f32_to_lds(const float* g, void* lds) {
  __attribute__((address_space(1))) int* gp =
      (__attribute__((address_space(1))) int*)(unsigned long long)(uintptr_t)g;
  __attribute__((address_space(3))) int* lp =
      (__attribute__((address_space(3))) int*)(unsigned int)(uintptr_t)lds;
  __builtin_amdgcn_global_load_async_to_lds_b32(gp, lp, 0, 0);
}
#if __has_builtin(__builtin_amdgcn_s_wait_asynccnt)
#define WAIT_ASYNC() __builtin_amdgcn_s_wait_asynccnt(0)
#else
#define WAIT_ASYNC() asm volatile("s_wait_asynccnt 0x0" ::: "memory")
#endif
#endif

__device__ __forceinline__ float act_apply(float v, int act) {
  if (act == 1) return v / (1.f + __expf(-v));                                // SiLU
  if (act == 2) return 0.5f * v * (1.f + erff(v * 0.70710678118654752f));     // exact GELU
  return v;
}

// ----------------------------------------------------------------------------
// Generic batched GEMM:  Out[m,n] = epilogue( alpha * sum_k A[m,k]*B[k,n] )
// Strided A/B/Out/Res so one kernel covers x@W^T linears, q@k^T, attn@v and
// W@X (1x1 conv / im2col 3x3 conv) forms. Block tile 128x64, K-stage 64
// (= 2 WMMA steps), double-buffered LDS, 8 waves; each wave owns a 32x32
// patch = 2x2 fragments -> 8 static v_wmma per loop body.
// Staging is branch-free: interior tiles take an unguarded path; edge tiles
// use clamped addresses + select (unconditional loads, no EXEC masking).
// ----------------------------------------------------------------------------
__global__ __launch_bounds__(256)
void gemm_bf16_wmma(const float* __restrict__ A, long long a_bs, int a_rs, int a_cs,
                    const float* __restrict__ Bm, long long b_bs, int b_rs, int b_cs,
                    float* __restrict__ Out, long long o_bs, int o_rs, int o_cs,
                    const float* __restrict__ bias, int bias_per_n,
                    const float* __restrict__ scale,            // per-M (fused BN), may be null
                    const float* __restrict__ Res, long long r_bs, int r_rs, int r_cs,
                    int M, int N, int K, float alpha, int act)
{
  __shared__ unsigned int AsU[2][BM * BK / 2];   // bf16 pairs, 16 KB each
  __shared__ unsigned int BsU[2][BK * BN / 2];   // bf16 pairs,  8 KB each

  const int tid  = threadIdx.x;
  const int lane = tid & 31;
  const int wave = tid >> 5;
  const int wM   = wave >> 1;            // 0..3 -> 32-row band
  const int wN   = wave & 1;             // 0..1 -> 32-col band
  const int bM0  = blockIdx.x * BM;
  const int bN0  = blockIdx.y * BN;
  const int bat  = blockIdx.z;

  A   += (long long)bat * a_bs;
  Bm  += (long long)bat * b_bs;
  Out += (long long)bat * o_bs;
  if (Res) Res += (long long)bat * r_bs;

  const bool mfull = (bM0 + BM <= M);
  const bool nfull = (bN0 + BN <= N);

  v8f acc[2][2] = {};

  auto stage = [&](int buf, int k0) {
    const bool kfull = (k0 + BK <= K);
    // ---- A tile: BM*BK/2 = 4096 bf16-pairs, 16 per thread -----------------
    float a0[16], a1[16];
    if (mfull & kfull) {
#pragma unroll
      for (int i = 0; i < 16; ++i) {
        int e = i * 256 + tid;
        int m = e >> 5, k = (e & 31) << 1;               // BK/2 = 32 pairs/row
        const float* p = A + (long long)(bM0 + m) * a_rs + (long long)(k0 + k) * a_cs;
        a0[i] = p[0]; a1[i] = p[a_cs];
      }
    } else {
#pragma unroll
      for (int i = 0; i < 16; ++i) {
        int e = i * 256 + tid;
        int m = e >> 5, k = (e & 31) << 1;
        int gm = bM0 + m, gk = k0 + k;
        bool i0 = (gm < M) & (gk < K);
        bool i1 = (gm < M) & (gk + 1 < K);
        float v0 = A[(long long)(i0 ? gm : 0) * a_rs + (long long)(i0 ? gk : 0) * a_cs];
        float v1 = A[(long long)(i1 ? gm : 0) * a_rs + (long long)(i1 ? gk + 1 : 0) * a_cs];
        a0[i] = i0 ? v0 : 0.f;
        a1[i] = i1 ? v1 : 0.f;
      }
    }
#pragma unroll
    for (int i = 0; i < 16; ++i) {
      union { __bf16 h[2]; unsigned int u; } pk;
      pk.h[0] = (__bf16)a0[i]; pk.h[1] = (__bf16)a1[i];
      AsU[buf][i * 256 + tid] = pk.u;                    // pair idx == elem/2
    }
    // ---- B tile: BK*BN/2 = 2048 bf16-pairs (pair along n), 8 per thread ---
    float b0[8], b1[8];
    if (nfull & kfull) {
#pragma unroll
      for (int i = 0; i < 8; ++i) {
        int e = i * 256 + tid;
        int k = e >> 5, n = (e & 31) << 1;               // BN/2 = 32 pairs/row
        const float* p = Bm + (long long)(k0 + k) * b_rs + (long long)(bN0 + n) * b_cs;
        b0[i] = p[0]; b1[i] = p[b_cs];
      }
    } else {
#pragma unroll
      for (int i = 0; i < 8; ++i) {
        int e = i * 256 + tid;
        int k = e >> 5, n = (e & 31) << 1;
        int gk = k0 + k, gn = bN0 + n;
        bool i0 = (gk < K) & (gn < N);
        bool i1 = (gk < K) & (gn + 1 < N);
        float v0 = Bm[(long long)(i0 ? gk : 0) * b_rs + (long long)(i0 ? gn : 0) * b_cs];
        float v1 = Bm[(long long)(i1 ? gk : 0) * b_rs + (long long)(i1 ? gn + 1 : 0) * b_cs];
        b0[i] = i0 ? v0 : 0.f;
        b1[i] = i1 ? v1 : 0.f;
      }
    }
#pragma unroll
    for (int i = 0; i < 8; ++i) {
      union { __bf16 h[2]; unsigned int u; } pk;
      pk.h[0] = (__bf16)b0[i]; pk.h[1] = (__bf16)b1[i];
      BsU[buf][i * 256 + tid] = pk.u;
    }
  };

  stage(0, 0);
  __syncthreads();

  const int nk   = (K + BK - 1) / BK;
  const int half = lane >> 4;
  const int r    = lane & 15;

  for (int kt = 0; kt < nk; ++kt) {
    const int buf = kt & 1;
    if (kt + 1 < nk) stage(buf ^ 1, (kt + 1) * BK);
    if (kt + 2 < nk) {
      int pm = bM0 + (tid >> 1); if (pm >= M) pm = M - 1;
      __builtin_prefetch(&A[(long long)pm * a_rs + (long long)((kt + 2) * BK) * a_cs], 0, 1);
    }

    const __bf16* Asb = (const __bf16*)AsU[buf];
    const __bf16* Bsb = (const __bf16*)BsU[buf];

#pragma unroll
    for (int ks = 0; ks < BK; ks += 32) {
      // A fragment, ISA 16-bit 16x32 layout:
      //  lanes 0-15 : halves 0..7 = K 0..7,  8..15 = K 16..23
      //  lanes16-31 : halves 0..7 = K 8..15, 8..15 = K 24..31
      v16bf afr[2], bfr[2];
#pragma unroll
      for (int wm = 0; wm < 2; ++wm) {
        const __bf16* ap = Asb + (wM * 32 + wm * 16 + r) * BK + ks + half * 8;
        v8bf lo = *(const v8bf*)(ap);
        v8bf hi = *(const v8bf*)(ap + 16);
        v16bf a;
#pragma unroll
        for (int i = 0; i < 8; ++i) { a[i] = lo[i]; a[8 + i] = hi[i]; }
        afr[wm] = a;
      }
      // B fragment, 32x16: lane%16 = N column, halves follow A's K order.
#pragma unroll
      for (int wn = 0; wn < 2; ++wn) {
        v16bf b;
#pragma unroll
        for (int i = 0; i < 16; ++i)
          b[i] = Bsb[(ks + half * 8 + (i & 7) + (i >> 3) * 16) * BN + wN * 32 + wn * 16 + r];
        bfr[wn] = b;
      }

#pragma unroll
      for (int wm = 0; wm < 2; ++wm)
#pragma unroll
        for (int wn = 0; wn < 2; ++wn)
          acc[wm][wn] = __builtin_amdgcn_wmma_f32_16x16x32_bf16(
              false, afr[wm], false, bfr[wn], (short)0, acc[wm][wn], false, false);
    }
    __syncthreads();
  }

  // Epilogue. C/D layout: lanes 0-15 VGPR j -> (M=j, N=lane);
  // lanes 16-31 VGPR j -> (M=8+j, N=lane-16).
  const bool ofull = mfull & nfull;
#pragma unroll
  for (int wm = 0; wm < 2; ++wm)
#pragma unroll
    for (int wn = 0; wn < 2; ++wn)
#pragma unroll
      for (int j = 0; j < 8; ++j) {
        int gm = bM0 + wM * 32 + wm * 16 + half * 8 + j;
        int gn = bN0 + wN * 32 + wn * 16 + r;
        if (ofull || (gm < M && gn < N)) {
          float v = acc[wm][wn][j] * alpha;
          if (scale) v *= scale[gm];
          if (bias)  v += bias[bias_per_n ? gn : gm];
          v = act_apply(v, act);
          if (Res) v += Res[(long long)gm * r_rs + (long long)gn * r_cs];
          Out[(long long)gm * o_rs + (long long)gn * o_cs] = v;
        }
      }
}

// ---------------------------------------------------------------------------
// LayerNorm: one wave32 per row (shfl_xor tree reductions).
// ---------------------------------------------------------------------------
__global__ __launch_bounds__(256)
void layernorm_rows(const float* __restrict__ X, float* __restrict__ Y,
                    const float* __restrict__ g, const float* __restrict__ b,
                    int rows, int L)
{
  int row  = blockIdx.x * (blockDim.x >> 5) + (threadIdx.x >> 5);
  int lane = threadIdx.x & 31;
  if (row >= rows) return;
  const float* x = X + (long long)row * L;
  float* y = Y + (long long)row * L;
  float s = 0.f, ss = 0.f;
  for (int i = lane; i < L; i += 32) { float v = x[i]; s += v; ss += v * v; }
  for (int o = 16; o; o >>= 1) { s += __shfl_xor(s, o, 32); ss += __shfl_xor(ss, o, 32); }
  float mu   = s / (float)L;
  float var  = ss / (float)L - mu * mu;
  float rstd = rsqrtf(var + 1e-5f);
  for (int i = lane; i < L; i += 32) y[i] = (x[i] - mu) * rstd * g[i] + b[i];
}

// ---------------------------------------------------------------------------
// Row softmax in place: one wave32 per row.
// ---------------------------------------------------------------------------
__global__ __launch_bounds__(256)
void softmax_rows(float* __restrict__ X, int rows, int L)
{
  int row  = blockIdx.x * (blockDim.x >> 5) + (threadIdx.x >> 5);
  int lane = threadIdx.x & 31;
  if (row >= rows) return;
  float* x = X + (long long)row * L;
  float m = -3.4e38f;
  for (int i = lane; i < L; i += 32) m = fmaxf(m, x[i]);
  for (int o = 16; o; o >>= 1) m = fmaxf(m, __shfl_xor(m, o, 32));
  float s = 0.f;
  for (int i = lane; i < L; i += 32) { float e = __expf(x[i] - m); x[i] = e; s += e; }
  for (int o = 16; o; o >>= 1) s += __shfl_xor(s, o, 32);
  float inv = 1.f / s;
  for (int i = lane; i < L; i += 32) x[i] *= inv;
}

// ---------------------------------------------------------------------------
// Batched 2D transpose: out[b][c][r] = in[b][r][c]   (32x32 LDS tile).
// Interior tiles stream straight into LDS with gfx1250 async loads
// (ASYNCcnt-tracked, no VGPR round trip).
// ---------------------------------------------------------------------------
__global__ __launch_bounds__(256)
void transpose2d(const float* __restrict__ in, float* __restrict__ out,
                 int B, int R, int C)
{
  __shared__ float tile[32][33];
  int b  = blockIdx.z;
  int r0 = blockIdx.y * 32, c0 = blockIdx.x * 32;
  const float* ib = in  + (long long)b * R * C;
  float*       ob = out + (long long)b * R * C;
  int tx = threadIdx.x & 31, ty = threadIdx.x >> 5;   // 32 x 8
  bool full = (r0 + 32 <= R) && (c0 + 32 <= C);
#if HAVE_ASYNC_LDS
  if (full) {
#pragma unroll
    for (int i = 0; i < 32; i += 8)
      async_load_f32_to_lds(&ib[(long long)(r0 + ty + i) * C + (c0 + tx)],
                            &tile[ty + i][tx]);
    WAIT_ASYNC();
    __syncthreads();
  } else
#endif
  {
    for (int i = 0; i < 32; i += 8) {
      int r = r0 + ty + i, c = c0 + tx;
      tile[ty + i][tx] = (r < R && c < C) ? ib[(long long)r * C + c] : 0.f;
    }
    __syncthreads();
  }
  (void)full;
  for (int i = 0; i < 32; i += 8) {
    int c = c0 + ty + i, r = r0 + tx;
    if (c < C && r < R) ob[(long long)c * R + r] = tile[tx][ty + i];
  }
}

// ---------------------------------------------------------------------------
// out[b][r][c] += in[b][c][r]   (the two c2f cross-stream adds)
// ---------------------------------------------------------------------------
__global__ __launch_bounds__(256)
void add_transposed(float* __restrict__ out, const float* __restrict__ in,
                    int B, int R, int C)
{
  long long total = (long long)B * R * C;
  for (long long idx = (long long)blockIdx.x * blockDim.x + threadIdx.x;
       idx < total; idx += (long long)gridDim.x * blockDim.x) {
    long long b = idx / ((long long)R * C);
    int rem = (int)(idx - b * (long long)R * C);
    int r = rem / C, c = rem % C;
    out[idx] += in[b * (long long)R * C + (long long)c * R + r];
  }
}

// ---------------------------------------------------------------------------
// im2col for 3x3 pad-1 conv over a 20x20 grid: Col[b][ci*9+tap][p]
// ---------------------------------------------------------------------------
__global__ __launch_bounds__(256)
void im2col3x3(const float* __restrict__ X, long long x_bs,
               float* __restrict__ Col, int B, int Ci)
{
  long long total = (long long)B * Ci * 9 * 400;
  for (long long idx = (long long)blockIdx.x * blockDim.x + threadIdx.x;
       idx < total; idx += (long long)gridDim.x * blockDim.x) {
    int p = (int)(idx % 400);
    long long t = idx / 400;
    int k = (int)(t % (Ci * 9));
    long long b = t / (Ci * 9);
    int ci = k / 9, tap = k % 9;
    int y = p / 20 + tap / 3 - 1;
    int x = p % 20 + tap % 3 - 1;
    bool in = (y >= 0) & (y < 20) & (x >= 0) & (x < 20);
    int yy = in ? y : 0, xx = in ? x : 0;
    float v = X[b * x_bs + (long long)ci * 400 + yy * 20 + xx];
    Col[idx] = in ? v : 0.f;
  }
}

// ---------------------------------------------------------------------------
// Depthwise 3x3 conv (c2f2 bottleneck cv2, groups==channels), fused BN+SiLU.
// ---------------------------------------------------------------------------
__global__ __launch_bounds__(256)
void dwconv3x3(const float* __restrict__ X, long long x_bs,
               const float* __restrict__ W, const float* __restrict__ s,
               const float* __restrict__ bia,
               float* __restrict__ O, long long o_bs, int B, int Ch)
{
  long long total = (long long)B * Ch * 400;
  for (long long idx = (long long)blockIdx.x * blockDim.x + threadIdx.x;
       idx < total; idx += (long long)gridDim.x * blockDim.x) {
    int p = (int)(idx % 400);
    long long t = idx / 400;
    int ch = (int)(t % Ch);
    long long b = t / Ch;
    int y0 = p / 20, x0 = p % 20;
    const float* w = W + ch * 9;
    const float* xb = X + b * x_bs + (long long)ch * 400;
    float acc = 0.f;
#pragma unroll
    for (int ky = 0; ky < 3; ++ky)
#pragma unroll
      for (int kx = 0; kx < 3; ++kx) {
        int y = y0 + ky - 1, x = x0 + kx - 1;
        if (y >= 0 && y < 20 && x >= 0 && x < 20)
          acc += w[ky * 3 + kx] * xb[y * 20 + x];
      }
    float v = acc * s[ch] + bia[ch];
    O[b * o_bs + (long long)ch * 400 + p] = v / (1.f + __expf(-v));
  }
}

// ============================================================================
// Host side
// ============================================================================
namespace {

struct ConvP { const float *b, *s, *w; };
struct LinP  { const float *b, *w; };
struct LnP   { const float *b, *g; };
struct C2fP  { ConvP cv1, cv2; ConvP m_cv1[2], m_cv2[2]; };

struct PW {  // parameter walker over d_in in jax tree_flatten (sorted-key) order
  void* const* din; int n; int i; int skip;
  const float* next() {
    if (i == skip) ++i;
    const float* p = (i >= 0 && i < n) ? (const float*)din[i] : nullptr;
    ++i; return p;
  }
  ConvP conv() { ConvP c; c.b = next(); c.s = next(); c.w = next(); return c; }
  LinP  lin()  { LinP l;  l.b = next(); l.w = next(); return l; }
  LnP   ln()   { LnP l;   l.b = next(); l.g = next(); return l; }
  C2fP  c2f()  { C2fP c; c.cv1 = conv(); c.cv2 = conv();
                 for (int k = 0; k < 2; ++k) { c.m_cv1[k] = conv(); c.m_cv2[k] = conv(); }
                 return c; }
};

inline void gemm(hipStream_t st,
                 const float* A, long long abs_, int ars, int acs,
                 const float* Bm, long long bbs, int brs, int bcs,
                 float* O, long long obs, int ors, int ocs,
                 const float* bias, int bias_per_n, const float* scale,
                 const float* Res, long long rbs, int rrs, int rcs,
                 int M, int N, int K, float alpha, int act, int batches)
{
  dim3 g((M + BM - 1) / BM, (N + BN - 1) / BN, batches);
  gemm_bf16_wmma<<<g, dim3(256), 0, st>>>(A, abs_, ars, acs, Bm, bbs, brs, bcs,
                                          O, obs, ors, ocs, bias, bias_per_n, scale,
                                          Res, rbs, rrs, rcs, M, N, K, alpha, act);
}

inline void ln_launch(hipStream_t st, const float* X, float* Y, const LnP& p, int rows, int L) {
  layernorm_rows<<<(rows + 7) / 8, 256, 0, st>>>(X, Y, p.g, p.b, rows, L);
}

// c2f block on (B,256,20,20) input with batch stride in_bs.
void run_c2f(hipStream_t st, const float* in, long long in_bs, const C2fP& p,
             bool depthwise, float* cat, float* tbuf, float* col, float* out)
{
  const int Bn = 32;
  // cv1: 256->256 1x1, BN+SiLU -> cat[ch 0..255]
  gemm(st, p.cv1.w, 0, 256, 1, in, in_bs, 400, 1,
       cat, 204800, 400, 1, p.cv1.b, 0, p.cv1.s,
       nullptr, 0, 0, 0, 256, 400, 256, 1.f, 1, Bn);
  for (int i = 0; i < 2; ++i) {
    const float* cur = cat + (long long)(128 + i * 128) * 400;   // y2, m1
    float*       dst = cat + (long long)(256 + i * 128) * 400;   // m1, m2 slots
    // bottleneck cv1: 128->128 1x1 -> tbuf
    gemm(st, p.m_cv1[i].w, 0, 128, 1, cur, 204800, 400, 1,
         tbuf, 51200, 400, 1, p.m_cv1[i].b, 0, p.m_cv1[i].s,
         nullptr, 0, 0, 0, 128, 400, 128, 1.f, 1, Bn);
    if (depthwise) {
      dwconv3x3<<<(32 * 128 * 400 + 255) / 256, 256, 0, st>>>(
          tbuf, 51200, p.m_cv2[i].w, p.m_cv2[i].s, p.m_cv2[i].b, dst, 204800, Bn, 128);
    } else {
      long long tot = (long long)32 * 1152 * 400;
      im2col3x3<<<(int)((tot + 255) / 256), 256, 0, st>>>(tbuf, 51200, col, Bn, 128);
      gemm(st, p.m_cv2[i].w, 0, 1152, 1, col, 460800, 400, 1,
           dst, 204800, 400, 1, p.m_cv2[i].b, 0, p.m_cv2[i].s,
           nullptr, 0, 0, 0, 128, 400, 1152, 1.f, 1, Bn);
    }
  }
  // cv2: 512->256 1x1 over concat
  gemm(st, p.cv2.w, 0, 512, 1, cat, 204800, 400, 1,
       out, 102400, 400, 1, p.cv2.b, 0, p.cv2.s,
       nullptr, 0, 0, 0, 256, 400, 512, 1.f, 1, Bn);
}

} // namespace

extern "C" void kernel_launch(void* const* d_in, const int* in_sizes, int n_in,
                              void* d_out, int out_size, void* d_ws, size_t ws_size,
                              hipStream_t stream)
{
  const int Bn = 32, H = 8;
  const long long SZ = (long long)Bn * 400 * 256;      // 3,276,800 elems

  // Locate x (2,B,N,C) = 6,553,600 elems; remaining inputs are params leaves
  // in jax tree_flatten (sorted-key) order.
  int xi = 0;
  for (int i = 0; i < n_in; ++i) if ((long long)in_sizes[i] == 2 * SZ) { xi = i; break; }
  const float* x0 = (const float*)d_in[xi];            // sa input (B,400,256)
  const float* x1 = x0 + SZ;                           // (B,400,256), pre-swap

  PW pw; pw.din = d_in; pw.n = n_in; pw.i = (xi == 0) ? 1 : 0; pw.skip = xi;
  // Sorted top-level keys: icmp1, icmp2, ln_in_sa, ln_in_se, ln_out_sa,
  //                        ln_out_se, mlp1, mlp2, sa1, sa2
  C2fP i1a = pw.c2f(); C2fP i1b = pw.c2f(); (void)i1a; (void)i1b;  // icmp1: DEAD
  C2fP c2f1 = pw.c2f(); C2fP c2f2 = pw.c2f();                      // icmp2
  LnP ln_in_sa = pw.ln(), ln_in_se = pw.ln(), ln_out_sa = pw.ln(), ln_out_se = pw.ln();
  LinP m1_fc1 = pw.lin(), m1_fc2 = pw.lin();
  LinP m2_fc1 = pw.lin(), m2_fc2 = pw.lin();
  LinP s1_kv = pw.lin(), s1_pr = pw.lin(), s1_q = pw.lin();
  LinP s2_kv = pw.lin(), s2_pr = pw.lin(), s2_q = pw.lin();

  // Workspace carve-out (~322 MB f32).
  char* wp = (char*)d_ws;
  auto alloc = [&](long long elems) {
    float* p = (float*)wp; wp += ((elems * 4 + 255) & ~255LL); return p;
  };
  float* se     = alloc(SZ);                    // (B,256,400)
  float* sa_n   = alloc(SZ);
  float* se_n   = alloc(SZ);
  float* qb     = alloc(SZ);                    // q (both branches)
  float* kvb    = alloc(2 * SZ);                // kv (both branches)
  float* ctx    = alloc(SZ);
  float* scores = alloc((long long)Bn * 400 * 400);
  float* sa_buf = alloc(SZ);
  float* se_buf = alloc(SZ);
  float* lnt    = alloc(SZ);
  float* hidden = alloc((long long)Bn * 400 * 1024);   // == B*256*1600
  float* x1r    = alloc(SZ);
  float* catb   = alloc((long long)Bn * 512 * 400);
  float* tbuf   = alloc((long long)Bn * 128 * 400);
  float* colb   = alloc((long long)Bn * 1152 * 400);
  float* a_buf  = alloc(SZ);
  float* s_buf  = alloc(SZ);
  (void)ws_size; (void)out_size;

  float* out0 = (float*)d_out;        // (B,400,256)
  float* out1 = out0 + SZ;            // swapaxes(se_out) target (B,400,256)

  // ---- se = swapaxes(x[1],1,2); input LayerNorms --------------------------
  transpose2d<<<dim3(8, 13, Bn), 256, 0, stream>>>(x1, se, Bn, 400, 256);
  ln_launch(stream, x0, sa_n, ln_in_sa, Bn * 400, 256);
  ln_launch(stream, se, se_n, ln_in_se, Bn * 256, 400);
  // icmp1(sa_n, se_n) is unused in the reference -> eliminated.

  // ---- self_attn branch 1: tokens=400, dim=256, dh=32 ---------------------
  gemm(stream, sa_n, 102400, 256, 1, s1_q.w, 0, 1, 256,
       qb, 102400, 256, 1, s1_q.b, 1, nullptr, nullptr, 0, 0, 0,
       400, 256, 256, 1.f, 0, Bn);
  gemm(stream, sa_n, 102400, 256, 1, s1_kv.w, 0, 1, 256,
       kvb, 204800, 512, 1, s1_kv.b, 1, nullptr, nullptr, 0, 0, 0,
       400, 512, 256, 1.f, 0, Bn);
  for (int h = 0; h < H; ++h) {
    gemm(stream, qb + h * 32, 102400, 256, 1, kvb + h * 32, 204800, 1, 512,
         scores, 160000, 400, 1, nullptr, 0, nullptr, nullptr, 0, 0, 0,
         400, 400, 32, 0.1767766952966369f, 0, Bn);
    softmax_rows<<<(Bn * 400 + 7) / 8, 256, 0, stream>>>(scores, Bn * 400, 400);
    gemm(stream, scores, 160000, 400, 1, kvb + 256 + h * 32, 204800, 512, 1,
         ctx + h * 32, 102400, 256, 1, nullptr, 0, nullptr, nullptr, 0, 0, 0,
         400, 32, 400, 1.f, 0, Bn);
  }
  gemm(stream, ctx, 102400, 256, 1, s1_pr.w, 0, 1, 256,
       sa_buf, 102400, 256, 1, s1_pr.b, 1, nullptr,
       x0, 102400, 256, 1, 400, 256, 256, 1.f, 0, Bn);     // + residual sa

  // ---- self_attn branch 2: tokens=256, dim=400, dh=50 ---------------------
  gemm(stream, se_n, 102400, 400, 1, s2_q.w, 0, 1, 400,
       qb, 102400, 400, 1, s2_q.b, 1, nullptr, nullptr, 0, 0, 0,
       256, 400, 400, 1.f, 0, Bn);
  gemm(stream, se_n, 102400, 400, 1, s2_kv.w, 0, 1, 400,
       kvb, 204800, 800, 1, s2_kv.b, 1, nullptr, nullptr, 0, 0, 0,
       256, 800, 400, 1.f, 0, Bn);
  for (int h = 0; h < H; ++h) {
    gemm(stream, qb + h * 50, 102400, 400, 1, kvb + h * 50, 204800, 1, 800,
         scores, 65536, 256, 1, nullptr, 0, nullptr, nullptr, 0, 0, 0,
         256, 256, 50, 0.1414213562373095f, 0, Bn);
    softmax_rows<<<(Bn * 256 + 7) / 8, 256, 0, stream>>>(scores, Bn * 256, 256);
    gemm(stream, scores, 65536, 256, 1, kvb + 400 + h * 50, 204800, 800, 1,
         ctx + h * 50, 102400, 400, 1, nullptr, 0, nullptr, nullptr, 0, 0, 0,
         256, 50, 256, 1.f, 0, Bn);
  }
  gemm(stream, ctx, 102400, 400, 1, s2_pr.w, 0, 1, 400,
       se_buf, 102400, 400, 1, s2_pr.b, 1, nullptr,
       se, 102400, 400, 1, 256, 400, 400, 1.f, 0, Bn);     // + residual se

  // ---- icmp2: a = c2f(sa^T, c2f1[3x3 dense]); s = c2f(se, c2f2[depthwise])
  transpose2d<<<dim3(8, 13, Bn), 256, 0, stream>>>(sa_buf, x1r, Bn, 400, 256);
  run_c2f(stream, x1r, 102400, c2f1, /*depthwise=*/false, catb, tbuf, colb, a_buf);
  run_c2f(stream, se_buf, 102400, c2f2, /*depthwise=*/true, catb, tbuf, colb, s_buf);

  // ---- MLP sa: LN -> fc1(GELU) -> fc2 (+res sa_buf) -> out0 ---------------
  ln_launch(stream, sa_buf, lnt, ln_out_sa, Bn * 400, 256);
  gemm(stream, lnt, 102400, 256, 1, m1_fc1.w, 0, 1, 256,
       hidden, 409600, 1024, 1, m1_fc1.b, 1, nullptr, nullptr, 0, 0, 0,
       400, 1024, 256, 1.f, 2, Bn);
  gemm(stream, hidden, 409600, 1024, 1, m1_fc2.w, 0, 1, 1024,
       out0, 102400, 256, 1, m1_fc2.b, 1, nullptr,
       sa_buf, 102400, 256, 1, 400, 256, 1024, 1.f, 0, Bn);

  // ---- MLP se: LN -> fc1(GELU) -> fc2 (+res se_buf) -> out1 (transposed) --
  ln_launch(stream, se_buf, lnt, ln_out_se, Bn * 256, 400);
  gemm(stream, lnt, 102400, 400, 1, m2_fc1.w, 0, 1, 400,
       hidden, 409600, 1600, 1, m2_fc1.b, 1, nullptr, nullptr, 0, 0, 0,
       256, 1600, 400, 1.f, 2, Bn);
  gemm(stream, hidden, 409600, 1600, 1, m2_fc2.w, 0, 1, 1600,
       out1, 102400, /*o_rs=*/1, /*o_cs=*/256, m2_fc2.b, 1, nullptr,
       se_buf, 102400, 400, 1, 256, 400, 1600, 1.f, 0, Bn);

  // ---- cross-stream c2f additions ----------------------------------------
  int nadd = (int)((SZ + 255) / 256);
  add_transposed<<<nadd, 256, 0, stream>>>(out0, s_buf, Bn, 400, 256);  // + s^T
  add_transposed<<<nadd, 256, 0, stream>>>(out1, a_buf, Bn, 400, 256);  // + a^T
}
